// CostVolumeAttention17_66546223284247
// MI455X (gfx1250) — matrix operand order repaired and checked by reading, not compiled, not run on hardware
//
#include <hip/hip_runtime.h>
#include <hip/hip_bf16.h>

typedef __attribute__((ext_vector_type(16))) __bf16 v16bf;
typedef __attribute__((ext_vector_type(8)))  __bf16 v8bf;
typedef __attribute__((ext_vector_type(8)))  float  v8f;

#define T_LEN   8
#define HEADS   8
#define HDIM    16
#define CDIM    128
#define HSZ     56
#define HW2     3136      // 56*56
#define BATCH   2
#define BT_TOT  128       // BATCH*HEADS*T_LEN
#define NPB     25088     // T_LEN*HW2
#define NROWS   50176     // BATCH*NPB
#define NWIN    8192      // BT_TOT*64

static __device__ __forceinline__ v8f wmma_bf16(v16bf a, v16bf b, v8f c) {
  return __builtin_amdgcn_wmma_f32_16x16x32_bf16(false, a, false, b, (short)0, c, false, false);
}

// --- CDNA5 async global->LDS copy (ASYNCcnt path, ISA 15.18 op 98) -----------
static __device__ __forceinline__ void async_copy_b128(void* lds_ptr, const void* gptr) {
  unsigned lds_off = (unsigned)(unsigned long long)lds_ptr;   // low 32 bits = LDS offset
  asm volatile("global_load_async_to_lds_b128 %0, %1, off"
               :: "v"(lds_off), "v"(gptr) : "memory");
}
static __device__ __forceinline__ void wait_async0() {
  asm volatile("s_wait_asynccnt 0x0" ::: "memory");
}

// --- fragment builders -------------------------------------------------------
// A (16x32 bf16): lane holds row M=lane%16; e    -> K=kc*32+16*half+e (16 contiguous)
// B (32x16 bf16): lane holds col N=lane%16; e    -> K=kc*32+16*half+e (16 contiguous)
static __device__ __forceinline__ v16bf cat8(v8bf lo, v8bf hi) {
  return __builtin_shufflevector(lo, hi, 0,1,2,3,4,5,6,7,8,9,10,11,12,13,14,15);
}
static __device__ __forceinline__ v16bf afrag_bf(const __bf16* row, int kc, int half) {
  v8bf lo = *(const v8bf*)(row + kc * 32 + 16 * half);
  v8bf hi = *(const v8bf*)(row + kc * 32 + 16 * half + 8);
  return cat8(lo, hi);
}
static __device__ __forceinline__ v16bf bfrag_bf(const __bf16* row, int kc, int half) {
  v8bf lo = *(const v8bf*)(row + kc * 32 + 16 * half);
  v8bf hi = *(const v8bf*)(row + kc * 32 + 16 * half + 8);
  return cat8(lo, hi);
}
static __device__ __forceinline__ v16bf afrag_f32(const float* row, int kc, int half) {
  const float* p0 = row + kc * 32 + 16 * half;
  const float* p1 = p0 + 8;
  v16bf r;
#pragma unroll
  for (int e = 0; e < 8; ++e) { r[e] = (__bf16)p0[e]; r[e + 8] = (__bf16)p1[e]; }
  return r;
}
static __device__ __forceinline__ v16bf bfrag_g(const float* p) {
  v16bf r;
#pragma unroll
  for (int e = 0; e < 16; ++e) r[e] = (__bf16)p[e];
  return r;
}

// ---------------------------------------------------------------------------
// Kernel 1: QKV GEMM.  out[m,n] = sum_k x[m,k]*qkv_w[n,k], M=50176 N=384 K=128.
// A-tile (16x128 f32, contiguous 8KB) staged via async global->LDS.
// ---------------------------------------------------------------------------
__global__ __launch_bounds__(128) void qkv_gemm_kernel(
    const float* __restrict__ x, const float* __restrict__ qkv_w,
    float* __restrict__ qbuf, float* __restrict__ kbuf, float* __restrict__ vbuf) {
  __shared__ alignas(16) float AsF[16 * 132];
  const int mtile = blockIdx.x;
  const int tid = threadIdx.x;

  const float* src = x + (size_t)mtile * 16 * CDIM;
#pragma unroll
  for (int it = 0; it < 4; ++it) {
    int idx = tid + it * 128;                 // 512 x b128 = 8KB tile
    int r = idx >> 5, c = (idx & 31) * 4;
    async_copy_b128(&AsF[r * 132 + c], src + idx * 4);
  }
  wait_async0();
  __syncthreads();

  const int wave = tid >> 5, lane = tid & 31;
  const int lm = lane & 15, half = lane >> 4;

  v16bf afrag[4];
#pragma unroll
  for (int kc = 0; kc < 4; ++kc) afrag[kc] = afrag_f32(&AsF[lm * 132], kc, half);

  __builtin_prefetch(qkv_w + (size_t)wave * 16 * CDIM, 0, 3);

  for (int nt = wave; nt < 24; nt += 4) {
    const int ncol = nt * 16 + lm;
    const float* wrow = qkv_w + (size_t)ncol * CDIM;
    v8f acc = {};
#pragma unroll
    for (int kc = 0; kc < 4; ++kc)
      acc = wmma_bf16(afrag[kc], bfrag_g(wrow + kc * 32 + 16 * half), acc);

    const int which = ncol >> 7;              // 0=q 1=k 2=v
    const int head  = (ncol >> 4) & 7;
    const int c     = ncol & 15;
    float* dst = (which == 0) ? qbuf : ((which == 1) ? kbuf : vbuf);
#pragma unroll
    for (int r = 0; r < 8; ++r) {
      int m  = mtile * 16 + r + 8 * half;
      int b  = m / NPB, rem = m % NPB;
      int t  = rem / HW2, hw = rem % HW2;
      int bt = (b * HEADS + head) * T_LEN + t;
      dst[((size_t)bt * HW2 + hw) * HDIM + c] = acc[r];
    }
  }
}

// ---------------------------------------------------------------------------
// Kernel 2: L2-normalize q in place; kp = l2norm(k shifted one frame forward).
// ---------------------------------------------------------------------------
__global__ __launch_bounds__(256) void norm_shift_kernel(
    float* __restrict__ qbuf, const float* __restrict__ kbuf, float* __restrict__ kp) {
  int i = blockIdx.x * 256 + threadIdx.x;
  if (i >= BT_TOT * HW2) return;
  int bt = i / HW2, hw = i % HW2;

  float tmp[16];
  {
    float* qv = qbuf + (size_t)i * HDIM;
    float ss = 0.f;
#pragma unroll
    for (int c = 0; c < 16; ++c) { tmp[c] = qv[c]; ss += tmp[c] * tmp[c]; }
    float inv = 1.f / fmaxf(sqrtf(ss), 1e-12f);
#pragma unroll
    for (int c = 0; c < 16; ++c) qv[c] = tmp[c] * inv;
  }
  {
    int t = bt & 7;
    int bts = (t == 7) ? bt : bt + 1;         // k_post = concat(k[:,1:], k[:,-1:])
    const float* kv = kbuf + ((size_t)bts * HW2 + hw) * HDIM;
    float ss = 0.f;
#pragma unroll
    for (int c = 0; c < 16; ++c) { tmp[c] = kv[c]; ss += tmp[c] * tmp[c]; }
    float inv = 1.f / fmaxf(sqrtf(ss), 1e-12f);
    float* ko = kp + (size_t)i * HDIM;
#pragma unroll
    for (int c = 0; c < 16; ++c) ko[c] = tmp[c] * inv;
  }
}

// ---------------------------------------------------------------------------
// Kernel 3: cost volume (49 shifted unit-vector dots) written in dilated-window
// layout corrw[widx][p][d] so the attention kernel reads one contiguous block.
// ---------------------------------------------------------------------------
__global__ __launch_bounds__(256) void corr_kernel(
    const float* __restrict__ qp, const float* __restrict__ kp,
    float* __restrict__ corrw) {
  int i = blockIdx.x * 256 + threadIdx.x;
  if (i >= BT_TOT * HW2) return;
  int bt = i / HW2, hw = i % HW2;
  int h = hw / HSZ, w = hw % HSZ;

  float q[16];
  const float* qv = qp + (size_t)i * HDIM;
#pragma unroll
  for (int c = 0; c < 16; ++c) q[c] = qv[c];

  int inh = h & 7, iws = h >> 3;              // h = iws*8 + inh (torch reshape)
  int inw = w & 7, jws = w >> 3;
  int widx = bt * 64 + inh * 8 + inw;
  int p = iws * 7 + jws;
  float* dst = corrw + ((size_t)widx * 49 + p) * 49;

  for (int du = -3; du <= 3; ++du)
    for (int dv = -3; dv <= 3; ++dv) {
      int hh = h + du, ww = w + dv;
      float s = 0.f;
      if (hh >= 0 && hh < HSZ && ww >= 0 && ww < HSZ) {
        const float* kv = kp + ((size_t)bt * HW2 + hh * HSZ + ww) * HDIM;
#pragma unroll
        for (int c = 0; c < 16; ++c) s += q[c] * kv[c];
      }
      *dst++ = s;
    }
}

// ---------------------------------------------------------------------------
// Kernel 4: fused window attention (motion + appearance), 2 waves / window.
// Corr window (9.6KB contiguous) staged via async global->LDS; all GEMMs WMMA.
// ---------------------------------------------------------------------------
__global__ __launch_bounds__(64) void win_attn_kernel(
    const float* __restrict__ corrw, const float* __restrict__ pe1,
    const float* __restrict__ qp,    const float* __restrict__ pe2,
    const float* __restrict__ vbuf,  float* __restrict__ x2) {
  __shared__ alignas(16) float  CWF[2432];     // raw 49x49 corr window (async staged)
  __shared__ alignas(16) __bf16 CW[64 * 72];   // corr + pe1, padded to 64x64
  __shared__ alignas(16) float  Sb[64 * 65];   // logits
  __shared__ alignas(16) __bf16 ATT[64 * 72];  // softmaxed attention
  __shared__ alignas(16) __bf16 VWt[16 * 72];  // v window, TRANSPOSED [c][p]
  __shared__ alignas(16) __bf16 QW[64 * 32];   // q window + pe2, K padded to 32
  __shared__ alignas(16) float  OUTM[64 * 16]; // motion branch result

  const int widx = blockIdx.x;
  const int bt = widx >> 6, wl = widx & 63;
  const int inh = wl >> 3, inw = wl & 7;
  const int tid = threadIdx.x;

  // async-stage the contiguous corr window: 2404 floats (601 x b128)
  const float* cwsrc = corrw + (size_t)widx * 2401;
  for (int it = 0; it < 10; ++it) {
    int idx = tid + it * 64;
    if (idx < 601) async_copy_b128(&CWF[idx * 4], cwsrc + idx * 4);
  }

  // gather v / q windows (dilated sampling -> not contiguous; regular loads)
  for (int idx = tid; idx < 64 * 16; idx += 64) {
    int p = idx >> 4, c = idx & 15;
    float v = 0.f, qv = 0.f;
    if (p < 49) {
      int iws = p / 7, jws = p % 7;
      int hw = (iws * 8 + inh) * HSZ + (jws * 8 + inw);
      size_t base = ((size_t)bt * HW2 + hw) * HDIM + c;
      v  = vbuf[base];
      qv = qp[base] + pe2[p * HDIM + c];
    }
    VWt[c * 72 + p] = (__bf16)v;
    QW[p * 32 + c] = (__bf16)qv;
    QW[p * 32 + 16 + c] = (__bf16)0.f;
  }
  wait_async0();
  __syncthreads();

  for (int idx = tid; idx < 64 * 64; idx += 64) {
    int p = idx >> 6, d = idx & 63;
    float v = 0.f;
    if (p < 49 && d < 49) v = CWF[p * 49 + d] + pe1[p * 49 + d];
    CW[p * 72 + d] = (__bf16)v;
  }
  __syncthreads();

  const int wave = tid >> 5, lane = tid & 31;
  const int lm = lane & 15, half = lane >> 4;

  // S = CW @ CW^T  (64x64x64, 16 tiles over 2 waves)
  for (int tix = wave; tix < 16; tix += 2) {
    int i = tix >> 2, j = tix & 3;
    v8f acc = {};
#pragma unroll
    for (int kc = 0; kc < 2; ++kc)
      acc = wmma_bf16(afrag_bf(&CW[(i * 16 + lm) * 72], kc, half),
                      bfrag_bf(&CW[(j * 16 + lm) * 72], kc, half), acc);
#pragma unroll
    for (int r = 0; r < 8; ++r)
      Sb[(i * 16 + r + 8 * half) * 65 + j * 16 + lm] = acc[r];
  }
  __syncthreads();

  // softmax rows 0..48 -> ATT
  {
    int r = tid;
    float mx = -1e30f, inv = 0.f;
    if (r < 49) {
      for (int d = 0; d < 49; ++d) mx = fmaxf(mx, Sb[r * 65 + d]);
      float sum = 0.f;
      for (int d = 0; d < 49; ++d) sum += __expf(Sb[r * 65 + d] - mx);
      inv = 1.f / sum;
    }
    for (int d = 0; d < 64; ++d) {
      float v = (r < 49 && d < 49) ? __expf(Sb[r * 65 + d] - mx) * inv : 0.f;
      ATT[r * 72 + d] = (__bf16)v;
    }
  }
  __syncthreads();

  // OUTM = ATT @ VW  (64x16x64), B fragments contiguous via transposed VWt
  for (int i = wave; i < 4; i += 2) {
    v8f acc = {};
#pragma unroll
    for (int kc = 0; kc < 2; ++kc)
      acc = wmma_bf16(afrag_bf(&ATT[(i * 16 + lm) * 72], kc, half),
                      bfrag_bf(&VWt[lm * 72], kc, half), acc);
#pragma unroll
    for (int r = 0; r < 8; ++r)
      OUTM[(i * 16 + r + 8 * half) * 16 + lm] = acc[r];
  }
  __syncthreads();

  // S2 = QW @ QW^T  (64x64x32, single K chunk)
  for (int tix = wave; tix < 16; tix += 2) {
    int i = tix >> 2, j = tix & 3;
    v8f acc = {};
    acc = wmma_bf16(afrag_bf(&QW[(i * 16 + lm) * 32], 0, half),
                    bfrag_bf(&QW[(j * 16 + lm) * 32], 0, half), acc);
#pragma unroll
    for (int r = 0; r < 8; ++r)
      Sb[(i * 16 + r + 8 * half) * 65 + j * 16 + lm] = acc[r];
  }
  __syncthreads();

  {
    int r = tid;
    float mx = -1e30f, inv = 0.f;
    if (r < 49) {
      for (int d = 0; d < 49; ++d) mx = fmaxf(mx, Sb[r * 65 + d]);
      float sum = 0.f;
      for (int d = 0; d < 49; ++d) sum += __expf(Sb[r * 65 + d] - mx);
      inv = 1.f / sum;
    }
    for (int d = 0; d < 64; ++d) {
      float v = (r < 49 && d < 49) ? __expf(Sb[r * 65 + d] - mx) * inv : 0.f;
      ATT[r * 72 + d] = (__bf16)v;
    }
  }
  __syncthreads();

  // app = ATT @ VW ; out = 0.5*(app + motion) scattered to (B,N,C)
  const int b_ = bt >> 6, head = (bt >> 3) & 7, t = bt & 7;
  for (int i = wave; i < 4; i += 2) {
    v8f acc = {};
#pragma unroll
    for (int kc = 0; kc < 2; ++kc)
      acc = wmma_bf16(afrag_bf(&ATT[(i * 16 + lm) * 72], kc, half),
                      bfrag_bf(&VWt[lm * 72], kc, half), acc);
#pragma unroll
    for (int r = 0; r < 8; ++r) {
      int p = i * 16 + r + 8 * half;
      if (p < 49) {
        int iws = p / 7, jws = p % 7;
        int hw = (iws * 8 + inh) * HSZ + (jws * 8 + inw);
        int row = b_ * NPB + t * HW2 + hw;
        int col = head * HDIM + lm;
        x2[(size_t)row * CDIM + col] = 0.5f * (OUTM[p * 16 + lm] + acc[r]);
      }
    }
  }
}

// ---------------------------------------------------------------------------
// Kernel 5: output projection. out[m,n] = sum_k x2[m,k]*proj_w[n,k] + proj_b[n]
// ---------------------------------------------------------------------------
__global__ __launch_bounds__(128) void proj_gemm_kernel(
    const float* __restrict__ x2, const float* __restrict__ pw,
    const float* __restrict__ pb, float* __restrict__ out) {
  __shared__ alignas(16) float AsF[16 * 132];
  const int mtile = blockIdx.x;
  const int tid = threadIdx.x;

  const float* src = x2 + (size_t)mtile * 16 * CDIM;
#pragma unroll
  for (int it = 0; it < 4; ++it) {
    int idx = tid + it * 128;
    int r = idx >> 5, c = (idx & 31) * 4;
    async_copy_b128(&AsF[r * 132 + c], src + idx * 4);
  }
  wait_async0();
  __syncthreads();

  const int wave = tid >> 5, lane = tid & 31;
  const int lm = lane & 15, half = lane >> 4;

  v16bf afrag[4];
#pragma unroll
  for (int kc = 0; kc < 4; ++kc) afrag[kc] = afrag_f32(&AsF[lm * 132], kc, half);

  for (int nt = wave; nt < 8; nt += 4) {
    const int ncol = nt * 16 + lm;
    const float* wrow = pw + (size_t)ncol * CDIM;
    v8f acc = {};
#pragma unroll
    for (int kc = 0; kc < 4; ++kc)
      acc = wmma_bf16(afrag[kc], bfrag_g(wrow + kc * 32 + 16 * half), acc);

    const float bias = pb[ncol];
#pragma unroll
    for (int r = 0; r < 8; ++r) {
      int m = mtile * 16 + r + 8 * half;
      out[(size_t)m * CDIM + ncol] = acc[r] + bias;
    }
  }
}

// ---------------------------------------------------------------------------
extern "C" void kernel_launch(void* const* d_in, const int* in_sizes, int n_in,
                              void* d_out, int out_size, void* d_ws, size_t ws_size,
                              hipStream_t stream) {
  const float* x      = (const float*)d_in[0];
  const float* qkv_w  = (const float*)d_in[1];
  const float* proj_w = (const float*)d_in[2];
  const float* proj_b = (const float*)d_in[3];
  const float* pe1    = (const float*)d_in[4];
  const float* pe2    = (const float*)d_in[5];
  float* out = (float*)d_out;

  float* ws = (float*)d_ws;
  const size_t SZQ = (size_t)BT_TOT * HW2 * HDIM;   // 6,422,528 floats
  float* qbuf  = ws;
  float* kbuf  = ws + SZQ;
  float* vbuf  = ws + 2 * SZQ;
  float* kpbuf = ws + 3 * SZQ;
  float* corrw = ws + 4 * SZQ;                      // NWIN*49*49 floats
  float* x2    = corrw + (size_t)NWIN * 49 * 49;    // NROWS*128 floats

  qkv_gemm_kernel<<<dim3(NROWS / 16), dim3(128), 0, stream>>>(x, qkv_w, qbuf, kbuf, vbuf);
  norm_shift_kernel<<<dim3((BT_TOT * HW2) / 256), dim3(256), 0, stream>>>(qbuf, kbuf, kpbuf);
  corr_kernel<<<dim3((BT_TOT * HW2) / 256), dim3(256), 0, stream>>>(qbuf, kpbuf, corrw);
  win_attn_kernel<<<dim3(NWIN), dim3(64), 0, stream>>>(corrw, pe1, qbuf, pe2, vbuf, x2);
  proj_gemm_kernel<<<dim3(NROWS / 16), dim3(128), 0, stream>>>(x2, proj_w, proj_b, out);
}
